// SignDenoiser_66941360275995
// MI455X (gfx1250) — matrix operand order repaired and checked by reading, not compiled, not run on hardware
//
#include <hip/hip_runtime.h>
#include <math.h>

// ---------------------------------------------------------------------------
// MI455X / gfx1250 implementation.
//  * All GEMMs (X @ W.T + b) run on v_wmma_f32_16x16x32_f16, f32 accumulate.
//    Weights are pre-converted once per launch into an f16 arena (halves
//    weight-side HBM/L2 traffic; removes per-tile cvt work).  Activations are
//    converted f32->f16 while staging to LDS.  Block tile 64(M) x 64(N) with
//    8 wave32; each wave computes 16x32 C (2 WMMA per K-step, A frag reused).
//    LDS layouts are arranged so fragments load as ds_load_b128.
//  * Epilogues (relu / softplus / sigmoid-gate) fused into the GEMM store.
//  * Mamba selective scan fused into one kernel: thread = (b, d) channel,
//    16 states in registers, 128 serial steps; never materializes the
//    reference's 128MB dA/dBx tensors.
// ---------------------------------------------------------------------------

typedef __attribute__((ext_vector_type(16))) _Float16 v16h;
typedef __attribute__((ext_vector_type(8)))  _Float16 v8h;
typedef __attribute__((ext_vector_type(8)))  float    v8f;

union F16Frag { v16h v; v8h h[2]; };

enum { EPI_NONE = 0, EPI_RELU = 1, EPI_SOFTPLUS = 2, EPI_GATE = 3 };

__device__ __forceinline__ float sigm(float x) { return 1.f / (1.f + expf(-x)); }

// ---------------------------------------------------------------------------
// Generic WMMA GEMM: C[M,N] = epi(sum_k A[m,k] * W[n,k] + bias[n])
//   A: f32 row-major, leading dim lda (may be pre-offset to a column slice).
//   W: f16 row-major (N x K), contiguous (pre-converted weight arena).
//   EPI_GATE: C = aux[m,n] * sigmoid(acc + bias)   (part_local gating)
// Block = 256 threads = 8 wave32. Block tile 64(M) x 64(N), K stepped by 32.
// Wave (wr, wc) handles rows [16wr,16wr+16), cols [32wc, 32wc+32).
// ---------------------------------------------------------------------------
__global__ __launch_bounds__(256)
void gemm_wmma_kernel(const float* __restrict__ A, int lda,
                      const _Float16* __restrict__ W,
                      const float* __restrict__ bias,
                      float* __restrict__ C, int ldc,
                      const float* __restrict__ aux, int ldaux,
                      int M, int N, int K, int epi)
{
    // Row pitch 40 halfs = 80B so the 16B fragment sub-runs stay 16B-aligned.
    __shared__ _Float16 As[64][40];   // As[m][k]
    __shared__ _Float16 Bs[64][40];   // Bs[n][k]  (k-contiguous per column)

    const int tid  = threadIdx.x;
    const int lane = tid & 31;
    const int wv   = tid >> 5;
    const int wr   = wv & 3;          // M sub-tile (0..3)
    const int wc   = wv >> 2;         // N sub-tile (0..1)
    const int m0   = blockIdx.x * 64;
    const int n0   = blockIdx.y * 64;

    v8f acc0 = {0.f, 0.f, 0.f, 0.f, 0.f, 0.f, 0.f, 0.f};
    v8f acc1 = {0.f, 0.f, 0.f, 0.f, 0.f, 0.f, 0.f, 0.f};

    const bool fullM = (m0 + 64 <= M);
    const bool fullN = (n0 + 64 <= N);

    for (int k0 = 0; k0 < K; k0 += 32) {
        const bool fullK = (k0 + 32 <= K);

        // ---- stage A tile (64 x 32), f32 -> f16: uniform fast path --------
        if (fullM && fullK) {
#pragma unroll
            for (int p = 0; p < 8; ++p) {
                int i = tid + p * 256;
                int r = i >> 5, c = i & 31;
                As[r][c] = (_Float16)A[(size_t)(m0 + r) * lda + (k0 + c)];
            }
        } else {
#pragma unroll
            for (int p = 0; p < 8; ++p) {
                int i = tid + p * 256;
                int r = i >> 5, c = i & 31;
                int gm = m0 + r, gk = k0 + c;
                float v = (gm < M && gk < K) ? A[(size_t)gm * lda + gk] : 0.f;
                As[r][c] = (_Float16)v;
            }
        }
        // ---- stage B tile (64 x 32) from f16 weights, [n][k] layout -------
        if (fullN && fullK) {
#pragma unroll
            for (int p = 0; p < 8; ++p) {
                int i = tid + p * 256;
                int n = i >> 5, c = i & 31;
                Bs[n][c] = W[(size_t)(n0 + n) * K + (k0 + c)];
            }
        } else {
#pragma unroll
            for (int p = 0; p < 8; ++p) {
                int i = tid + p * 256;
                int n = i >> 5, c = i & 31;
                int gn = n0 + n, gk = k0 + c;
                Bs[n][c] = (gn < N && gk < K) ? W[(size_t)gn * K + gk]
                                              : (_Float16)0.f;
            }
        }
        __syncthreads();

        // ---- fragments via ds_load_b128 ----------------------------------
        // A (16x32): lane<16 row m holds K {0..7,16..23}; lane>=16 holds
        //            K {8..15,24..31}  -> two contiguous 8-half runs.
        // B (32x16): lane<16 col n holds K {0..7,16..23}; lane>=16 holds
        //            K {8..15,24..31}  -> two contiguous 8-half runs.
        const int half = lane >> 4;
        const int mr   = (wr << 4) + (lane & 15);
        const int nc   = (wc << 5) + (lane & 15);
        F16Frag a, b0, b1;
        a.h[0]  = *(const v8h*)&As[mr][half * 8];
        a.h[1]  = *(const v8h*)&As[mr][16 + half * 8];
        b0.h[0] = *(const v8h*)&Bs[nc][half * 8];
        b0.h[1] = *(const v8h*)&Bs[nc][16 + half * 8];
        b1.h[0] = *(const v8h*)&Bs[nc + 16][half * 8];
        b1.h[1] = *(const v8h*)&Bs[nc + 16][16 + half * 8];

        acc0 = __builtin_amdgcn_wmma_f32_16x16x32_f16(
            false, a.v, false, b0.v, (short)0, acc0, false, false);
        acc1 = __builtin_amdgcn_wmma_f32_16x16x32_f16(
            false, a.v, false, b1.v, (short)0, acc1, false, false);
        __syncthreads();
    }

    // ---- store: C layout: VGPR r, lane l -> M = r + 8*(l>=16), N = l&15 ----
    const int ncA = n0 + (wc << 5) + (lane & 15);
    const int ncB = ncA + 16;
    const float bvA = (bias && ncA < N) ? bias[ncA] : 0.f;
    const float bvB = (bias && ncB < N) ? bias[ncB] : 0.f;
#pragma unroll
    for (int r = 0; r < 8; ++r) {
        int m = m0 + (wr << 4) + r + ((lane >> 4) << 3);
        if (m < M) {
            if (ncA < N) {
                float v = acc0[r] + bvA;
                if (epi == EPI_RELU)          v = v > 0.f ? v : 0.f;
                else if (epi == EPI_SOFTPLUS) v = (v > 20.f) ? v : log1pf(expf(v));
                else if (epi == EPI_GATE)     v = aux[(size_t)m * ldaux + ncA] * sigm(v);
                C[(size_t)m * ldc + ncA] = v;
            }
            if (ncB < N) {
                float v = acc1[r] + bvB;
                if (epi == EPI_RELU)          v = v > 0.f ? v : 0.f;
                else if (epi == EPI_SOFTPLUS) v = (v > 20.f) ? v : log1pf(expf(v));
                else if (epi == EPI_GATE)     v = aux[(size_t)m * ldaux + ncB] * sigm(v);
                C[(size_t)m * ldc + ncB] = v;
            }
        }
    }
}

// --------------------------- elementwise helpers ---------------------------

// f32 -> f16 weight conversion (once per launch per weight matrix)
__global__ void cvt_f16_kernel(const float* __restrict__ src,
                               _Float16* __restrict__ dst, long n)
{
    long i = (long)blockIdx.x * 256 + threadIdx.x;
    if (i < n) dst[i] = (_Float16)src[i];
}

__global__ void mask_rows_kernel(float* __restrict__ x, const int* __restrict__ mask,
                                 long total, int C)
{
    long i = (long)blockIdx.x * 256 + threadIdx.x;
    if (i < total) { if (!mask[i / C]) x[i] = 0.f; }
}

__global__ void mask_copy_y_kernel(const float* __restrict__ y, const int* __restrict__ ymask,
                                   float* __restrict__ ym, int Bn, int C)
{
    int i = blockIdx.x * 256 + threadIdx.x;
    if (i < Bn * C) ym[i] = ymask[i / C] ? y[i] : 0.f;
}

// Generic depthwise conv along L. in/out pointers may be column-offset.
__global__ void dwconv_kernel(const float* __restrict__ in, int ldin,
                              const float* __restrict__ w, int kw,
                              const float* __restrict__ bias,
                              float* __restrict__ out, int ldout,
                              int Bn, int Lin, int Lout, int C,
                              int stride, int padL, int act)
{
    long i = (long)blockIdx.x * 256 + threadIdx.x;
    long total = (long)Bn * Lout * C;
    if (i >= total) return;
    int c  = (int)(i % C);
    long t = i / C;
    int lo = (int)(t % Lout);
    int b  = (int)(t / Lout);
    float acc = bias ? bias[c] : 0.f;
    for (int k = 0; k < kw; ++k) {
        int pos = lo * stride - padL + k;
        if (pos >= 0 && pos < Lin)
            acc += w[c * kw + k] * in[((size_t)b * Lin + pos) * ldin + c];
    }
    if (act == 1) acc = acc * sigm(acc);   // silu
    out[((size_t)b * Lout + lo) * ldout + c] = acc;
}

// GroupNorm over (L, C/ng) per (batch, group), fused SiLU, in place.
// grid = Bn * ng, block = 256.
__global__ __launch_bounds__(256)
void groupnorm_silu_kernel(float* __restrict__ x, int ld,
                           const float* __restrict__ g, const float* __restrict__ b,
                           int Ln, int C, int ng)
{
    int grp = blockIdx.x % ng;
    int bb  = blockIdx.x / ng;
    int cpg = C / ng;
    int n   = Ln * cpg;
    __shared__ float s1[256], s2[256];
    float sum = 0.f, sq = 0.f;
    for (int i = threadIdx.x; i < n; i += 256) {
        int l = i / cpg, c = i % cpg;
        float v = x[((size_t)bb * Ln + l) * ld + grp * cpg + c];
        sum += v; sq += v * v;
    }
    s1[threadIdx.x] = sum; s2[threadIdx.x] = sq;
    __syncthreads();
    for (int st = 128; st > 0; st >>= 1) {
        if (threadIdx.x < st) { s1[threadIdx.x] += s1[threadIdx.x + st];
                                s2[threadIdx.x] += s2[threadIdx.x + st]; }
        __syncthreads();
    }
    float mu   = s1[0] / n;
    float var  = s2[0] / n - mu * mu;
    float rinv = rsqrtf(var + 1e-5f);
    for (int i = threadIdx.x; i < n; i += 256) {
        int l = i / cpg, c = i % cpg;
        size_t idx = ((size_t)bb * Ln + l) * ld + grp * cpg + c;
        int ch = grp * cpg + c;
        float v = (x[idx] - mu) * rinv * g[ch] + b[ch];
        x[idx] = v * sigm(v);
    }
}

// LayerNorm over C=256 contiguous; optional residual; grid = rows.
__global__ __launch_bounds__(256)
void layernorm_kernel(const float* __restrict__ x, const float* __restrict__ res,
                      float* __restrict__ out,
                      const float* __restrict__ g, const float* __restrict__ b, int C)
{
    int row = blockIdx.x, tid = threadIdx.x;
    float v = x[(size_t)row * C + tid];
    if (res) v += res[(size_t)row * C + tid];
    __shared__ float s1[256], s2[256];
    s1[tid] = v; s2[tid] = v * v;
    __syncthreads();
    for (int st = 128; st > 0; st >>= 1) {
        if (tid < st) { s1[tid] += s1[tid + st]; s2[tid] += s2[tid + st]; }
        __syncthreads();
    }
    float mu = s1[0] / C, var = s2[0] / C - mu * mu;
    float rinv = rsqrtf(var + 1e-5f);
    out[(size_t)row * C + tid] = (v - mu) * rinv * g[tid] + b[tid];
}

// x1 = x[:,1:] zero-padded to 512 rows.
__global__ void build_x1p_kernel(const float* __restrict__ h, float* __restrict__ x1p,
                                 int Bn, int C)
{
    long i = (long)blockIdx.x * 256 + threadIdx.x;
    long total = (long)Bn * 512 * C;
    if (i >= total) return;
    int c = (int)(i % C);
    long t = i / C;
    int l = (int)(t % 512), b = (int)(t / 512);
    x1p[i] = (l < 511) ? h[((size_t)b * 512 + l + 1) * C + c] : 0.f;
}

__global__ void copy_cols_kernel(const float* __restrict__ src, int lds_,
                                 float* __restrict__ dst, int ldd, long rows, int C)
{
    long i = (long)blockIdx.x * 256 + threadIdx.x;
    if (i >= rows * C) return;
    long r = i / C; int c = (int)(i % C);
    dst[(size_t)r * ldd + c] = src[(size_t)r * lds_ + c];
}

// mode 0: dst = y broadcast; mode 1: dst = y * sigmoid(yhat)
__global__ void bcast_gate_kernel(const float* __restrict__ y, const float* __restrict__ yhat,
                                  float* __restrict__ dst, int ldd,
                                  int Bn, int T, int C, int mode)
{
    long i = (long)blockIdx.x * 256 + threadIdx.x;
    long total = (long)Bn * T * C;
    if (i >= total) return;
    int c = (int)(i % C);
    long t = i / C;
    int tt = (int)(t % T), b = (int)(t / T);
    float v = y[(size_t)b * C + c];
    if (mode == 1) v *= sigm(yhat[((size_t)b * T + tt) * C + c]);
    dst[((size_t)b * T + tt) * ldd + c] = v;
}

// m_in = concat([x2 reversed, x2], axis=1)
__global__ void build_min_kernel(const float* __restrict__ x2, float* __restrict__ m_in,
                                 int Bn, int T, int C)
{
    long i = (long)blockIdx.x * 256 + threadIdx.x;
    long total = (long)Bn * 2 * T * C;
    if (i >= total) return;
    int c = (int)(i % C);
    long t = i / C;
    int l = (int)(t % (2 * T)), b = (int)(t / (2 * T));
    int src = (l < T) ? (T - 1 - l) : (l - T);
    m_in[i] = x2[((size_t)b * T + src) * C + c];
}

// Fused selective scan: thread = (b,d) channel, 16 states in registers.
// out = (scan + D*x) * silu(z).   dbl cols: dt 0..15 | B 16..31 | C 32..47.
__global__ __launch_bounds__(256)
void mamba_scan_kernel(const float* __restrict__ dt2, const float* __restrict__ xmc,
                       const float* __restrict__ dbl, const float* __restrict__ xz,
                       const float* __restrict__ A_log, const float* __restrict__ Dp,
                       float* __restrict__ out, int Ln)
{
    int d = (blockIdx.x & 1) * 256 + threadIdx.x;   // 0..511
    int b = blockIdx.x >> 1;
    float Ar[16], h[16];
#pragma unroll
    for (int s = 0; s < 16; ++s) { Ar[s] = -expf(A_log[d * 16 + s]); h[s] = 0.f; }
    float Dd = Dp[d];
    for (int l = 0; l < Ln; ++l) {
        size_t rb = (size_t)b * Ln + l;
        float dt = dt2[rb * 512 + d];
        float xv = xmc[rb * 512 + d];
        const float* row = dbl + rb * 48;
        float acc = 0.f;
#pragma unroll
        for (int s = 0; s < 16; ++s) {
            float da = expf(dt * Ar[s]);
            h[s] = da * h[s] + dt * row[16 + s] * xv;
            acc += h[s] * row[32 + s];
        }
        float z = xz[rb * 1024 + 512 + d];
        out[rb * 512 + d] = (acc + Dd * xv) * (z * sigm(z));
    }
}

// cat4[b,l, 0:256] = x1p[b,l]; cat4[b,l, 256:512] = nx2[b, 64 + l/8]
__global__ void build_cat4_kernel(const float* __restrict__ x1p, const float* __restrict__ nx2,
                                  float* __restrict__ cat4, int Bn, int C)
{
    long i = (long)blockIdx.x * 256 + threadIdx.x;
    long total = (long)Bn * 512 * 2 * C;
    if (i >= total) return;
    int c = (int)(i % (2 * C));
    long t = i / (2 * C);
    int l = (int)(t % 512), b = (int)(t / 512);
    float v;
    if (c < C) v = x1p[((size_t)b * 512 + l) * C + c];
    else       v = nx2[((size_t)b * 128 + 64 + (l >> 3)) * C + (c - C)];
    cat4[i] = v;
}

// out[b,0,:] = h[b,0,:]; out[b,l,:] = nx[b,l-1,:]  (l = 1..511)
__global__ void assemble_kernel(const float* __restrict__ h, const float* __restrict__ nx,
                                float* __restrict__ out, int Bn, int C)
{
    long i = (long)blockIdx.x * 256 + threadIdx.x;
    long total = (long)Bn * 512 * C;
    if (i >= total) return;
    int c = (int)(i % C);
    long t = i / C;
    int l = (int)(t % 512), b = (int)(t / 512);
    out[i] = (l == 0) ? h[((size_t)b * 512) * C + c]
                      : nx[((size_t)b * 512 + (l - 1)) * C + c];
}

// ---------------------------------------------------------------------------

extern "C" void kernel_launch(void* const* d_in, const int* in_sizes, int n_in,
                              void* d_out, int out_size, void* d_ws, size_t ws_size,
                              hipStream_t stream)
{
    (void)in_sizes; (void)n_in; (void)out_size;
    const int Bn = 32, Ln = 512;
    const int rows = Bn * Ln;                   // 16384

    const float* x     = (const float*)d_in[0];
    const int*   xmask = (const int*)  d_in[1];
    const float* y     = (const float*)d_in[2];
    const int*   ymask = (const int*)  d_in[3];

    // ---- workspace arena (phase-disjoint aliasing) ----
    float* ws = (float*)d_ws;
    size_t off = 0;
    auto alloc = [&](size_t n) { float* p = ws + off; off += n; return p; };
    float* H    = alloc((size_t)rows * 256);   // persistent hidden state
    float* A1   = alloc((size_t)rows * 256);   // cat / h-relu / pre-LN assemble
    float* A2   = alloc((size_t)rows * 256);   // pw scratch / x1 padded
    float* MRG  = alloc((size_t)rows * 256);   // merged / nx
    float* C4   = alloc((size_t)rows * 512);   // cat4; first 16MB doubles as xz
    float* NX1  = alloc((size_t)Bn * 64 * 256);
    float* YHAT = alloc((size_t)Bn * 64 * 256);
    float* CATA = alloc((size_t)Bn * 64 * 512);
    float* X2   = alloc((size_t)Bn * 64 * 256);
    float* MIN_ = alloc((size_t)Bn * 128 * 256);
    float* HN   = alloc((size_t)Bn * 128 * 256);
    float* XMC  = alloc((size_t)Bn * 128 * 512);
    float* DBL  = alloc((size_t)Bn * 128 * 48);
    float* DT2  = alloc((size_t)Bn * 128 * 512);
    float* SCY  = alloc((size_t)Bn * 128 * 512);
    float* NX2  = alloc((size_t)Bn * 128 * 256);
    float* YM   = alloc((size_t)Bn * 256);
    // f16 weight arena after the f32 region (~4.3M halfs)
    _Float16* WARE = (_Float16*)(ws + off);
    const size_t WCAP = 4400000;
    if (off * sizeof(float) + WCAP * sizeof(_Float16) > ws_size) return;
    float* XZ = C4;   // xz dead before cat4 is built

    auto g1 = [](long n) { return dim3((unsigned)((n + 255) / 256)); };

    // weight f32 -> f16 conversion (once per launch, stream-ordered)
    size_t woff = 0;
    auto wcvt = [&](const float* w, size_t n) {
        _Float16* dst = WARE + woff;
        woff += (n + 7) & ~(size_t)7;
        cvt_f16_kernel<<<g1((long)n), 256, 0, stream>>>(w, dst, (long)n);
        return (const _Float16*)dst;
    };

    // ---- parameter cursor (setup_inputs() insertion order, 312 leaves) ----
    int pi = 4;
    auto nxt = [&]() { return (const float*)d_in[pi++]; };

    const int msz[3] = {30, 45, 45}, hdz[3] = {64, 96, 96};

    const _Float16* ip_w[3]; const float* ip_b[3];
    for (int j = 0; j < 3; ++j) {
        ip_w[j] = wcvt(nxt(), (size_t)hdz[j] * msz[j]);
        ip_b[j] = nxt();
    }

    struct LmP  { const _Float16 *pw_w[3], *g_w[3];
                  const float *pw_b[3], *dw_w[3], *dw_b[3],
                              *gn_g[3], *gn_b[3], *g_b[3], *ln_g, *ln_b; };
    struct MbP  { const _Float16 *in_w, *xproj_w, *dt_w, *out_w;
                  const float *ln_g,*ln_b,*conv_w,*conv_b,*dt_b,*A_log,*Dp; };
    struct MixP { const _Float16 *lc_pw_w, *f_w, *fuse_w, *final_w;
                  const float *lc_pw_b,*lc_dw_w,*lc_dw_b; MbP mb;
                  const float *f_b,*fuse_b,*final_b,*ln_g,*ln_b; };

    auto readLm = [&](LmP& p) {
        for (int j = 0; j < 3; ++j) {
            p.pw_w[j] = wcvt(nxt(), (size_t)hdz[j] * hdz[j]);
            p.pw_b[j] = nxt(); p.dw_w[j] = nxt(); p.dw_b[j] = nxt();
            p.gn_g[j] = nxt(); p.gn_b[j] = nxt();
        }
        for (int j = 0; j < 3; ++j) {
            p.g_w[j] = wcvt(nxt(), (size_t)hdz[j] * 256);
            p.g_b[j] = nxt();
        }
        p.ln_g = nxt(); p.ln_b = nxt();
    };
    auto readMix = [&](MixP& p) {
        p.lc_pw_w = wcvt(nxt(), 256u * 256); p.lc_pw_b = nxt();
        p.lc_dw_w = nxt(); p.lc_dw_b = nxt();
        p.mb.ln_g = nxt(); p.mb.ln_b = nxt();
        p.mb.in_w = wcvt(nxt(), 1024u * 256);
        p.mb.conv_w = nxt(); p.mb.conv_b = nxt();
        p.mb.xproj_w = wcvt(nxt(), 48u * 512);
        p.mb.dt_w = wcvt(nxt(), 512u * 16);
        p.mb.dt_b = nxt(); p.mb.A_log = nxt(); p.mb.Dp = nxt();
        p.mb.out_w = wcvt(nxt(), 256u * 512);
        p.f_w = wcvt(nxt(), 256u * 512);     p.f_b = nxt();
        p.fuse_w = wcvt(nxt(), 256u * 512);  p.fuse_b = nxt();
        p.final_w = wcvt(nxt(), 256u * 512); p.final_b = nxt();
        p.ln_g = nxt(); p.ln_b = nxt();
    };

    LmP lm1[4], lm2[4]; MixP mix[4];
    for (int l = 0; l < 4; ++l) { readLm(lm1[l]); readMix(mix[l]); readLm(lm2[l]); }
    const _Float16* op_w[3]; const float* op_b[3];
    for (int j = 0; j < 3; ++j) {
        op_w[j] = wcvt(nxt(), (size_t)msz[j] * hdz[j]);
        op_b[j] = nxt();
    }

    auto gemm = [&](const float* A, int lda, const _Float16* W, const float* bias,
                    float* Cp, int ldc, int M, int N, int K, int epi,
                    const float* aux, int ldaux) {
        dim3 grid((M + 63) / 64, (N + 63) / 64);
        gemm_wmma_kernel<<<grid, 256, 0, stream>>>(A, lda, W, bias, Cp, ldc,
                                                   aux, ldaux, M, N, K, epi);
    };

    // masked y (constant across layers)
    mask_copy_y_kernel<<<g1((long)Bn * 256), 256, 0, stream>>>(y, ymask, YM, Bn, 256);

    // ---- in_proj: x[...,ms slices] -> H columns ----
    {
        int xo = 0, ho = 0;
        for (int j = 0; j < 3; ++j) {
            gemm(x + xo, 120, ip_w[j], ip_b[j], H + ho, 256,
                 rows, hdz[j], msz[j], EPI_NONE, nullptr, 0);
            xo += msz[j]; ho += hdz[j];
        }
    }

    auto part_local = [&](const LmP& p) {
        mask_rows_kernel<<<g1((long)rows * 256), 256, 0, stream>>>(
            H, xmask, (long)rows * 256, 256);
        int o = 0;
        for (int j = 0; j < 3; ++j) {
            gemm(H + o, 256, p.pw_w[j], p.pw_b[j], A2 + o, 256,
                 rows, hdz[j], hdz[j], EPI_NONE, nullptr, 0);
            dwconv_kernel<<<g1((long)Bn * Ln * hdz[j]), 256, 0, stream>>>(
                A2 + o, 256, p.dw_w[j], 3, p.dw_b[j], A1 + o, 256,
                Bn, Ln, Ln, hdz[j], 1, 1, 0);
            groupnorm_silu_kernel<<<Bn * 8, 256, 0, stream>>>(
                A1 + o, 256, p.gn_g[j], p.gn_b[j], Ln, hdz[j], 8);
            o += hdz[j];
        }
        o = 0;
        for (int j = 0; j < 3; ++j) {   // merged = cat_part * sigmoid(cat @ gW + gb)
            gemm(A1, 256, p.g_w[j], p.g_b[j], MRG + o, 256,
                 rows, hdz[j], 256, EPI_GATE, A1 + o, 256);
            o += hdz[j];
        }
        layernorm_kernel<<<rows, 256, 0, stream>>>(H, MRG, H, p.ln_g, p.ln_b, 256);
    };

    auto mixed = [&](const MixP& p) {
        mask_rows_kernel<<<g1((long)rows * 256), 256, 0, stream>>>(
            H, xmask, (long)rows * 256, 256);
        build_x1p_kernel<<<g1((long)rows * 256), 256, 0, stream>>>(H, A2, Bn, 256);
        gemm(A2, 256, p.lc_pw_w, p.lc_pw_b, A1, 256, rows, 256, 256, EPI_RELU, nullptr, 0);
        dwconv_kernel<<<g1((long)Bn * 64 * 256), 256, 0, stream>>>(
            A1, 256, p.lc_dw_w, 8, p.lc_dw_b, NX1, 256, Bn, 512, 64, 256, 8, 0, 0);
        copy_cols_kernel<<<g1((long)Bn * 64 * 256), 256, 0, stream>>>(
            NX1, 256, CATA, 512, (long)Bn * 64, 256);
        bcast_gate_kernel<<<g1((long)Bn * 64 * 256), 256, 0, stream>>>(
            YM, nullptr, CATA + 256, 512, Bn, 64, 256, 0);
        gemm(CATA, 512, p.f_w, p.f_b, YHAT, 256, Bn * 64, 256, 512, EPI_NONE, nullptr, 0);
        bcast_gate_kernel<<<g1((long)Bn * 64 * 256), 256, 0, stream>>>(
            YM, YHAT, CATA + 256, 512, Bn, 64, 256, 1);
        gemm(CATA, 512, p.fuse_w, p.fuse_b, X2, 256, Bn * 64, 256, 512, EPI_NONE, nullptr, 0);
        build_min_kernel<<<g1((long)Bn * 128 * 256), 256, 0, stream>>>(X2, MIN_, Bn, 64, 256);
        // --- mamba block ---
        layernorm_kernel<<<Bn * 128, 256, 0, stream>>>(
            MIN_, nullptr, HN, p.mb.ln_g, p.mb.ln_b, 256);
        gemm(HN, 256, p.mb.in_w, nullptr, XZ, 1024, Bn * 128, 1024, 256, EPI_NONE, nullptr, 0);
        dwconv_kernel<<<g1((long)Bn * 128 * 512), 256, 0, stream>>>(
            XZ, 1024, p.mb.conv_w, 4, p.mb.conv_b, XMC, 512, Bn, 128, 128, 512, 1, 3, 1);
        gemm(XMC, 512, p.mb.xproj_w, nullptr, DBL, 48, Bn * 128, 48, 512, EPI_NONE, nullptr, 0);
        gemm(DBL, 48, p.mb.dt_w, p.mb.dt_b, DT2, 512, Bn * 128, 512, 16, EPI_SOFTPLUS, nullptr, 0);
        mamba_scan_kernel<<<Bn * 2, 256, 0, stream>>>(
            DT2, XMC, DBL, XZ, p.mb.A_log, p.mb.Dp, SCY, 128);
        gemm(SCY, 512, p.mb.out_w, nullptr, NX2, 256, Bn * 128, 256, 512, EPI_NONE, nullptr, 0);
        // --- fuse back to sequence ---
        build_cat4_kernel<<<g1((long)rows * 512), 256, 0, stream>>>(A2, NX2, C4, Bn, 256);
        gemm(C4, 512, p.final_w, p.final_b, MRG, 256, rows, 256, 512, EPI_NONE, nullptr, 0);
        assemble_kernel<<<g1((long)rows * 256), 256, 0, stream>>>(H, MRG, A1, Bn, 256);
        layernorm_kernel<<<rows, 256, 0, stream>>>(A1, nullptr, H, p.ln_g, p.ln_b, 256);
    };

    for (int l = 0; l < 4; ++l) {
        part_local(lm1[l]);
        mixed(mix[l]);
        part_local(lm2[l]);
    }

    // ---- out_proj: H column slices -> output columns ----
    {
        int ho = 0, oo = 0;
        float* out = (float*)d_out;
        for (int j = 0; j < 3; ++j) {
            gemm(H + ho, 256, op_w[j], op_b[j], out + oo, 120,
                 rows, msz[j], hdz[j], EPI_NONE, nullptr, 0);
            ho += hdz[j]; oo += msz[j];
        }
    }
}
